// GNUMAP2_47777216201257
// MI455X (gfx1250) — matrix operand order repaired and checked by reading, not compiled, not run on hardware
//
#include <hip/hip_runtime.h>
#include <hip/hip_bf16.h>

typedef __attribute__((ext_vector_type(16))) _Float16 v16h;
typedef __attribute__((ext_vector_type(8)))  float    v8f;

#define IN_DIM 128
#define HIDDEN 64
#define ALPHA_C 0.1520f
#define BETA_C  0.7900f

// ---------------- degree / normalization ----------------

__global__ void k_init_deg(float* __restrict__ deg, int N) {
    int i = blockIdx.x * blockDim.x + threadIdx.x;
    if (i < N) deg[i] = 1.0f;            // self-loop contributes 1
}

__global__ void k_accum_deg(const int* __restrict__ ei, int E, float* __restrict__ deg) {
    int e = blockIdx.x * blockDim.x + threadIdx.x;
    if (e < E) atomicAdd(&deg[ei[e]], 1.0f);   // rows = edge_index[0]
}

__global__ void k_dinv(const float* __restrict__ deg, float* __restrict__ dinv, int N) {
    int i = blockIdx.x * blockDim.x + threadIdx.x;
    if (i < N) dinv[i] = rsqrtf(deg[i]);       // deg >= 1 always
}

// ---------------- W1 pre-pack into WMMA B-fragment layout (f16) ----------------
// Fragment order: [kc][ct][lane][e], e=0..15 halves per lane (32B contiguous).
// Element (kc,ct,lane,e) = W1[(kc*32 + 8*(lane>>4) + (e&7) + 16*(e>>3)) * 64 + ct*16 + (lane&15)]

__global__ void k_pack_w1(const float* __restrict__ W1, _Float16* __restrict__ Bp) {
    int t = blockIdx.x * blockDim.x + threadIdx.x;
    if (t >= 4 * 4 * 32 * 16) return;
    int e    = t & 15;
    int lane = (t >> 4) & 31;
    int ct   = (t >> 9) & 3;
    int kc   = (t >> 11) & 3;
    int k = kc * 32 + 8 * (lane >> 4) + (e & 7) + 16 * (e >> 3);
    int n = ct * 16 + (lane & 15);
    Bp[t] = (_Float16)W1[k * HIDDEN + n];
}

// ---------------- GEMM1: H = X @ W1 via WMMA f16 -> f32 ----------------
// One wave computes a 16-row x 64-col tile of H:
// 4 column tiles x 4 K-chunks of v_wmma_f32_16x16x32_f16.
// A fragments converted in-register from f32 (4x b128 loads per chunk);
// B fragments read pre-packed (2x b128 loads per fragment).

__global__ __launch_bounds__(32) void k_gemm1_wmma(const float* __restrict__ X,
                                                   const _Float16* __restrict__ Bp,
                                                   float* __restrict__ H, int N) {
    const int tile = blockIdx.x;
    const int lane = threadIdx.x & 31;
    const int half = lane >> 4;
    const int l15  = lane & 15;

    int arow = tile * 16 + l15;
    int arow_clamped = (arow < N) ? arow : (N - 1);
    const float* xrow = X + (size_t)arow_clamped * IN_DIM;
    const v16h*  Bv   = (const v16h*)Bp;

    v8f acc[4];
    #pragma unroll
    for (int ct = 0; ct < 4; ++ct) acc[ct] = (v8f){};

    #pragma unroll
    for (int kc = 0; kc < 4; ++kc) {
        // ---- A fragment: K groups {kbase..kbase+7} and {kbase+16..kbase+23} ----
        const int kbaseA = kc * 32 + half * 8;
        float4 f0 = *reinterpret_cast<const float4*>(xrow + kbaseA);
        float4 f1 = *reinterpret_cast<const float4*>(xrow + kbaseA + 4);
        float4 f2 = *reinterpret_cast<const float4*>(xrow + kbaseA + 16);
        float4 f3 = *reinterpret_cast<const float4*>(xrow + kbaseA + 20);
        v16h a;
        a[0]  = (_Float16)f0.x;  a[1]  = (_Float16)f0.y;
        a[2]  = (_Float16)f0.z;  a[3]  = (_Float16)f0.w;
        a[4]  = (_Float16)f1.x;  a[5]  = (_Float16)f1.y;
        a[6]  = (_Float16)f1.z;  a[7]  = (_Float16)f1.w;
        a[8]  = (_Float16)f2.x;  a[9]  = (_Float16)f2.y;
        a[10] = (_Float16)f2.z;  a[11] = (_Float16)f2.w;
        a[12] = (_Float16)f3.x;  a[13] = (_Float16)f3.y;
        a[14] = (_Float16)f3.z;  a[15] = (_Float16)f3.w;

        // ---- B fragments: pre-packed, contiguous 32B per lane ----
        #pragma unroll
        for (int ct = 0; ct < 4; ++ct) {
            v16h b = Bv[(kc * 4 + ct) * 32 + lane];
            acc[ct] = __builtin_amdgcn_wmma_f32_16x16x32_f16(
                false, a, false, b, (short)0, acc[ct], false, false);
        }
    }

    // ---- store D fragments ----
    #pragma unroll
    for (int ct = 0; ct < 4; ++ct) {
        #pragma unroll
        for (int r = 0; r < 8; ++r) {
            int row = tile * 16 + r + 8 * half;
            if (row < N)
                H[(size_t)row * HIDDEN + ct * 16 + l15] = acc[ct][r];
        }
    }
}

// ---------------- layer-1 aggregation ----------------

__global__ void k_init_agg1(const float* __restrict__ H, const float* __restrict__ dinv,
                            const float* __restrict__ b1, float* __restrict__ agg1, int N) {
    int t = blockIdx.x * blockDim.x + threadIdx.x;
    if (t >= N * HIDDEN) return;
    int i = t >> 6, j = t & 63;
    float s = dinv[i] * dinv[i];                 // self-loop norm
    agg1[t] = fmaf(H[t], s, b1[j]);
}

__global__ void k_agg1_edges(const int* __restrict__ ei, int E,
                             const float* __restrict__ dinv,
                             const float* __restrict__ H, float* __restrict__ agg1) {
    long long tid = (long long)blockIdx.x * blockDim.x + threadIdx.x;
    if (tid >= (long long)E * 32) return;
    int e  = (int)(tid >> 5);                   // one edge per wave (broadcast loads)
    int j2 = (int)(tid & 31) * 2;
    int r = ei[e];
    int c = ei[E + e];
    float w = dinv[r] * dinv[c];
    float2 h = *reinterpret_cast<const float2*>(H + (size_t)c * HIDDEN + j2);
    atomicAdd(&agg1[(size_t)r * HIDDEN + j2],     h.x * w);
    atomicAdd(&agg1[(size_t)r * HIDDEN + j2 + 1], h.y * w);
}

// ---------------- layer 2: P = relu(agg1) @ W2 ; emb self-loop init ----------------

__global__ void k_gemm2_emb(const float* __restrict__ agg1, const float* __restrict__ W2,
                            const float* __restrict__ b2, const float* __restrict__ dinv,
                            float* __restrict__ P, float* __restrict__ emb, int N) {
    int i = blockIdx.x * blockDim.x + threadIdx.x;
    if (i >= N) return;
    float p0 = 0.f, p1 = 0.f;
    const float4* row = reinterpret_cast<const float4*>(agg1 + (size_t)i * HIDDEN);
    #pragma unroll
    for (int k4 = 0; k4 < HIDDEN / 4; ++k4) {
        float4 v = row[k4];
        float h0 = fmaxf(v.x, 0.f), h1 = fmaxf(v.y, 0.f);
        float h2 = fmaxf(v.z, 0.f), h3 = fmaxf(v.w, 0.f);
        int k = k4 * 4;
        p0 = fmaf(h0, W2[2 * k],     fmaf(h1, W2[2 * k + 2], fmaf(h2, W2[2 * k + 4], fmaf(h3, W2[2 * k + 6], p0))));
        p1 = fmaf(h0, W2[2 * k + 1], fmaf(h1, W2[2 * k + 3], fmaf(h2, W2[2 * k + 5], fmaf(h3, W2[2 * k + 7], p1))));
    }
    P[2 * i]     = p0;
    P[2 * i + 1] = p1;
    float s = dinv[i] * dinv[i];
    emb[2 * i]     = fmaf(p0, s, b2[0]);
    emb[2 * i + 1] = fmaf(p1, s, b2[1]);
}

__global__ void k_agg2_edges(const int* __restrict__ ei, int E,
                             const float* __restrict__ dinv,
                             const float* __restrict__ P, float* __restrict__ emb) {
    int e = blockIdx.x * blockDim.x + threadIdx.x;
    if (e >= E) return;
    int r = ei[e];
    int c = ei[E + e];
    float w = dinv[r] * dinv[c];
    atomicAdd(&emb[2 * r],     P[2 * c]     * w);
    atomicAdd(&emb[2 * r + 1], P[2 * c + 1] * w);
}

// ---------------- UMAP q over positive + negative pairs ----------------
// q = 1/(1 + a * d^(2b)) with d = sqrt(dd + eps)  ==>  1/(1 + a * (dd+eps)^b)

__global__ void k_q(const int* __restrict__ ei, const int* __restrict__ rneg,
                    const int* __restrict__ cneg, int E,
                    const float* __restrict__ emb, float* __restrict__ qout) {
    int t = blockIdx.x * blockDim.x + threadIdx.x;
    if (t >= 2 * E) return;
    int a, b;
    if (t < E) { a = ei[t];       b = ei[E + t]; }
    else       { a = rneg[t - E]; b = cneg[t - E]; }
    float dx = emb[2 * a]     - emb[2 * b];
    float dy = emb[2 * a + 1] - emb[2 * b + 1];
    float dd = fmaf(dx, dx, dy * dy) + 1e-12f;
    qout[t] = 1.0f / (1.0f + ALPHA_C * __powf(dd, BETA_C));
}

// ---------------- launch ----------------

extern "C" void kernel_launch(void* const* d_in, const int* in_sizes, int n_in,
                              void* d_out, int out_size, void* d_ws, size_t ws_size,
                              hipStream_t stream) {
    const float* feats = (const float*)d_in[0];
    const float* W1    = (const float*)d_in[1];
    const float* b1    = (const float*)d_in[2];
    const float* W2    = (const float*)d_in[3];
    const float* b2    = (const float*)d_in[4];
    const int*   ei    = (const int*)d_in[5];   // [2, E] : rows then cols
    const int*   rneg  = (const int*)d_in[6];
    const int*   cneg  = (const int*)d_in[7];

    const int N = in_sizes[0] / IN_DIM;
    const int E = in_sizes[6];

    float* ws   = (float*)d_ws;
    float* deg  = ws;
    float* dinv = deg  + N;
    float* H    = dinv + N;
    float* agg1 = H    + (size_t)N * HIDDEN;
    float* P    = agg1 + (size_t)N * HIDDEN;
    _Float16* Bp = (_Float16*)(P + (size_t)N * 2);   // 16 KB, 16B-aligned (32B if N is even)

    float* emb  = (float*)d_out;                // [N,2]
    float* qout = emb + (size_t)N * 2;          // [2E]

    const int B = 256;
    k_init_deg <<<(N + B - 1) / B, B, 0, stream>>>(deg, N);
    k_accum_deg<<<(E + B - 1) / B, B, 0, stream>>>(ei, E, deg);
    k_dinv     <<<(N + B - 1) / B, B, 0, stream>>>(deg, dinv, N);

    k_pack_w1<<<(4 * 4 * 32 * 16 + B - 1) / B, B, 0, stream>>>(W1, Bp);

    int nTiles = (N + 15) / 16;
    k_gemm1_wmma<<<nTiles, 32, 0, stream>>>(feats, Bp, H, N);

    long long a1threads = (long long)N * HIDDEN;
    k_init_agg1<<<(int)((a1threads + B - 1) / B), B, 0, stream>>>(H, dinv, b1, agg1, N);

    long long ethreads = (long long)E * 32;
    k_agg1_edges<<<(int)((ethreads + B - 1) / B), B, 0, stream>>>(ei, E, dinv, H, agg1);

    k_gemm2_emb<<<(N + B - 1) / B, B, 0, stream>>>(agg1, W2, b2, dinv, P, emb, N);
    k_agg2_edges<<<(E + B - 1) / B, B, 0, stream>>>(ei, E, dinv, P, emb);

    k_q<<<(2 * E + B - 1) / B, B, 0, stream>>>(ei, rneg, cneg, E, emb, qout);
}
